// RelationalGraphNeuralNetwork_10170482557020
// MI455X (gfx1250) — matrix operand order repaired and checked by reading, not compiled, not run on hardware
//
#include <hip/hip_runtime.h>
#include <hip/hip_bf16.h>
#include <math.h>

typedef __attribute__((ext_vector_type(16))) __bf16 v16bf;
typedef __attribute__((ext_vector_type(8)))  __bf16 v8bf;
typedef __attribute__((ext_vector_type(8)))  float  v8f;
typedef __attribute__((ext_vector_type(4)))  float  v4f;

#define SMOOTH_F 12.0f
#define EMB 64

#ifndef USE_TDM
#if defined(__has_builtin)
#if __has_builtin(__builtin_amdgcn_tensor_load_to_lds) && __has_builtin(__builtin_amdgcn_s_wait_tensorcnt)
#define USE_TDM 1
#else
#define USE_TDM 0
#endif
#else
#define USE_TDM 0
#endif
#endif

// Branchless mish via hardware exp/rcp:
// tanh(softplus(x)) = t(t+2)/(t(t+2)+2), t = e^x ;  for x>20, mish(x) ~= x
__device__ __forceinline__ float mish_f(float x) {
  float xc = fminf(x, 20.0f);
  float t  = __expf(xc);
  float u  = t * (t + 2.0f);
  float mv = xc * u * __builtin_amdgcn_rcpf(u + 2.0f);
  return (x > 20.0f) ? x : mv;
}

__device__ __forceinline__ void atomic_max_f32(float* p, float v) {
  __hip_atomic_fetch_max(p, v, __ATOMIC_RELAXED, __HIP_MEMORY_SCOPE_AGENT);
}
__device__ __forceinline__ void atomic_add_f32(float* p, float v) {
  __hip_atomic_fetch_add(p, v, __ATOMIC_RELAXED, __HIP_MEMORY_SCOPE_AGENT);
}

// ---------------- Tensor Data Mover helpers ----------------
#if USE_TDM
typedef __attribute__((ext_vector_type(4))) unsigned int u32x4;
typedef __attribute__((ext_vector_type(8))) int          i32x8;
typedef __attribute__((ext_vector_type(4))) int          i32x4;

// 1-D DMA: nbytes (multiple of 8) from global gsrc -> LDS byte offset lds_byte_off.
// D# group0: count=1 | lds_addr | global_addr[56:0] | type=2
// D# group1: data_size=8B, tensor_dim0=tile_dim0=n8, tensor_dim1=1, stride=n8
__device__ __forceinline__ void tdm_load_1d(const void* gsrc, unsigned lds_byte_off,
                                            unsigned nbytes) {
  unsigned long long ga = (unsigned long long)gsrc;
  unsigned n8 = nbytes >> 3;
  u32x4 g0;
  g0[0] = 1u;                                                 // count=1, user desc
  g0[1] = lds_byte_off;                                       // lds_addr (bytes)
  g0[2] = (unsigned)(ga & 0xffffffffu);                       // global_addr[31:0]
  g0[3] = (unsigned)((ga >> 32) & 0x01ffffffu) | (2u << 30);  // addr[56:32] | type=2
  i32x8 g1;
  g1[0] = (3 << 16);                          // workgroup_mask=0, data_size=8B
  g1[1] = (int)((n8 & 0xffffu) << 16);        // tensor_dim0[15:0]
  g1[2] = (int)((n8 >> 16) & 0xffffu) | (1 << 16);  // tensor_dim0[31:16] | tensor_dim1=1
  g1[3] = (int)((n8 & 0xffffu) << 16);        // tile_dim0 = n8
  g1[4] = 0;                                  // tile_dim1=0 (1-D), tile_dim2=0
  g1[5] = (int)n8;                            // tensor_dim0_stride[31:0]
  g1[6] = 0;
  g1[7] = 0;
  i32x4 z4 = {0, 0, 0, 0};
  i32x8 z8 = {0, 0, 0, 0, 0, 0, 0, 0};
  // clang-23 / amdgpu-toolchain 6-arg form: (g0, g1, g2, g3, g4, cpol)
  __builtin_amdgcn_tensor_load_to_lds(g0, g1, z4, z4, z8, 0);
}
#endif

// ---------------- small utility kernels ----------------

__global__ void k_convert_transpose(const float* __restrict__ W,
                                    unsigned short* __restrict__ Wt_,
                                    int K, int Nn) {
  __bf16* Wt = (__bf16*)Wt_;
  int i = blockIdx.x * blockDim.x + threadIdx.x;
  if (i < K * Nn) {
    int k = i / Nn, n = i - k * Nn;
    Wt[n * K + k] = (__bf16)W[i];   // store transposed [Nn][K]
  }
}

__global__ void k_init_agg(float* __restrict__ seg, float* __restrict__ sum, int n) {
  int i = blockIdx.x * blockDim.x + threadIdx.x;
  if (i < n) { seg[i] = -3.0e38f; sum[i] = 1e-16f; }
}

__global__ void k_fix_max(float* __restrict__ seg, int n) {
  int i = blockIdx.x * blockDim.x + threadIdx.x;
  if (i < n) { float v = seg[i]; seg[i] = (v < -1.0e30f) ? 0.0f : v; }
}

__global__ void k_final_msg(float* __restrict__ msg, const float* __restrict__ sum,
                            const float* __restrict__ seg, int n) {
  int i = blockIdx.x * blockDim.x + threadIdx.x;
  if (i < n) msg[i] = __logf(sum[i]) * (1.0f / SMOOTH_F) + seg[i];
}

// ---------------- WMMA fragment builders ----------------

// A fragment (16x32 bf16) from an fp32 LDS row; lane<16 -> K {0..7,16..23}, lane>=16 -> K {8..15,24..31}
__device__ __forceinline__ v16bf make_afrag_f32(const float* row, int k0) {
  v4f p0 = *(const v4f*)(row + k0);
  v4f p1 = *(const v4f*)(row + k0 + 4);
  v4f p2 = *(const v4f*)(row + k0 + 16);
  v4f p3 = *(const v4f*)(row + k0 + 20);
  v16bf a;
#pragma unroll
  for (int j = 0; j < 4; ++j) {
    a[j]      = (__bf16)p0[j];
    a[4 + j]  = (__bf16)p1[j];
    a[8 + j]  = (__bf16)p2[j];
    a[12 + j] = (__bf16)p3[j];
  }
  return a;
}

// A fragment from bf16 LDS row (hidden activations)
__device__ __forceinline__ v16bf make_afrag_bf16(const __bf16* row, int k0) {
  v8bf a0 = *(const v8bf*)(row + k0);
  v8bf a1 = *(const v8bf*)(row + k0 + 16);
  return __builtin_shufflevector(a0, a1, 0,1,2,3,4,5,6,7,8,9,10,11,12,13,14,15);
}

// B fragment (32x16 bf16) from transposed weights: wrow = &W[n*K]; K {kb..kb+7, kb+16..kb+23}
__device__ __forceinline__ v16bf make_bfrag(const __bf16* wrow, int kb) {
  v8bf b0 = *(const v8bf*)(wrow + kb);
  v8bf b1 = *(const v8bf*)(wrow + kb + 16);
  return __builtin_shufflevector(b0, b1, 0,1,2,3,4,5,6,7,8,9,10,11,12,13,14,15);
}

// ---------------- fused gather + residual MLP (+ scatter epilogue) ----------------
// MODE 0: relation pass A  -> atomic max into out0 = seg_max[N*64]
// MODE 1: relation pass B  -> atomic add exp(S*(y - exps_max)) into out0 = exps_sum
// MODE 2: update MLP: x = [msg | h] (WIN=128), out0 = h_next = h + mlp(x)
template<int WIN, int WHID, int WOUT, int MODE, int ARITY>
__global__ __launch_bounds__(128)
void fused_gemm(const float* __restrict__ hin,
                const float* __restrict__ msg,
                const int*   __restrict__ idx,
                int m,
                const unsigned short* __restrict__ WtI_,
                const float* __restrict__ bi,
                const unsigned short* __restrict__ WtO_,
                const float* __restrict__ bo,
                float* __restrict__ out0,
                const float* __restrict__ exps_max) {
  extern __shared__ char smem[];
  constexpr int XB = 64 * WIN * 4;
  constexpr int TB = 64 * WHID * 2;
  constexpr int WB = WHID * WIN * 2;
  float*  x_lds  = (float*)smem;                   // [64][WIN] fp32 (residual + GEMM1 A)
  __bf16* t_lds  = (__bf16*)(smem + XB);           // [64][WHID] bf16 hidden
  __bf16* wi_lds = (__bf16*)(smem + XB + TB);      // [WHID][WIN] transposed Wi
  __bf16* wo_lds = (__bf16*)(smem + XB + TB + WB); // [WOUT][WHID] transposed Wo

  const int tid  = threadIdx.x;
  const int lane = tid & 31;
  const int wave = tid >> 5;
  const int rowBase = blockIdx.x * 64;
  const int l15  = lane & 15;
  const int hl   = lane >> 4;     // 0: lanes 0-15, 1: lanes 16-31
  const int mbase = wave * 16;

  // ---- stage weights (bf16, pre-transposed) into LDS ----
#if USE_TDM
  if (wave == 0) {
    // Async Global->LDS DMA via Tensor Data Mover; overlaps with x-tile gather below.
    tdm_load_1d(WtI_, XB + TB, WHID * WIN * 2);
    tdm_load_1d(WtO_, XB + TB + WB, WOUT * WHID * 2);
  }
#else
  {
    const uint4* src = (const uint4*)WtI_;
    uint4* dst = (uint4*)wi_lds;
    int n = (WHID * WIN * 2) / 16;
    for (int i = tid; i < n; i += 128) dst[i] = src[i];
    src = (const uint4*)WtO_;
    dst = (uint4*)wo_lds;
    n = (WOUT * WHID * 2) / 16;
    for (int i = tid; i < n; i += 128) dst[i] = src[i];
  }
#endif

  // ---- stage x tile [64][WIN] fp32 (gather for relations; concat for update) ----
  {
    const int chunks = 64 * WIN / 4;
    for (int cc = tid; cc < chunks; cc += 128) {
      int r  = cc / (WIN / 4);
      int c4 = (cc - r * (WIN / 4)) * 4;
      int grow = rowBase + r;
      int gr = grow < m ? grow : m - 1;
      const float* src;
      if (MODE == 2) {
        src = (c4 < EMB) ? (msg + (size_t)gr * EMB + c4)
                         : (hin + (size_t)gr * EMB + (c4 - EMB));
      } else {
        int slot = c4 >> 6;
        int node = idx[gr * ARITY + slot];
        src = hin + (size_t)node * EMB + (c4 & 63);
      }
      *(v4f*)&x_lds[r * WIN + c4] = *(const v4f*)src;
    }
  }
#if USE_TDM
  if (wave == 0) __builtin_amdgcn_s_wait_tensorcnt(0);
#endif
  __syncthreads();

  // ---- GEMM1: hidden = mish(x @ Wi + bi) ----
  v8f acc[WHID / 16];
#pragma unroll
  for (int nt = 0; nt < WHID / 16; ++nt)
#pragma unroll
    for (int j = 0; j < 8; ++j) acc[nt][j] = 0.0f;

  const int arow = mbase + l15;
  for (int kt = 0; kt < WIN / 32; ++kt) {
    v16bf afrag = make_afrag_f32(&x_lds[arow * WIN], kt * 32 + hl * 8);
    int kb = kt * 32 + hl * 8;
#pragma unroll
    for (int nt = 0; nt < WHID / 16; ++nt) {
      v16bf bfrag = make_bfrag(&wi_lds[(nt * 16 + l15) * WIN], kb);
      acc[nt] = __builtin_amdgcn_wmma_f32_16x16x32_bf16(
          false, afrag, false, bfrag, (short)0, acc[nt], false, false);
    }
  }

  // epilogue 1: bias + mish -> bf16 hidden tile
#pragma unroll
  for (int nt = 0; nt < WHID / 16; ++nt) {
    int col = nt * 16 + l15;
    float bv = bi[col];
#pragma unroll
    for (int r = 0; r < 8; ++r) {
      int rl = mbase + hl * 8 + r;
      t_lds[rl * WHID + col] = (__bf16)mish_f(acc[nt][r] + bv);
    }
  }
  __syncthreads();

  // ---- GEMM2: y = hidden @ Wo + bo (+ residual) ----
  v8f acc2[WOUT / 16];
#pragma unroll
  for (int nt = 0; nt < WOUT / 16; ++nt)
#pragma unroll
    for (int j = 0; j < 8; ++j) acc2[nt][j] = 0.0f;

  for (int kt = 0; kt < WHID / 32; ++kt) {
    v16bf afrag = make_afrag_bf16(&t_lds[arow * WHID], kt * 32 + hl * 8);
    int kb = kt * 32 + hl * 8;
#pragma unroll
    for (int nt = 0; nt < WOUT / 16; ++nt) {
      v16bf bfrag = make_bfrag(&wo_lds[(nt * 16 + l15) * WHID], kb);
      acc2[nt] = __builtin_amdgcn_wmma_f32_16x16x32_bf16(
          false, afrag, false, bfrag, (short)0, acc2[nt], false, false);
    }
  }

  // ---- pass-specific epilogue ----
#pragma unroll
  for (int nt = 0; nt < WOUT / 16; ++nt) {
    int col = nt * 16 + l15;
    float bv = bo[col];
#pragma unroll
    for (int r = 0; r < 8; ++r) {
      int rl = mbase + hl * 8 + r;
      int grow = rowBase + rl;
      if (grow >= m) continue;
      float y = acc2[nt][r] + bv;
      if (MODE == 2) {
        // h_next = h + mlp(x); h is second half of x tile
        out0[(size_t)grow * EMB + col] = x_lds[rl * WIN + EMB + col] + y;
      } else {
        y += x_lds[rl * WIN + col];   // residual relation MLP
        int node = idx[grow * ARITY + (col >> 6)];
        size_t a = (size_t)node * EMB + (col & 63);
        if (MODE == 0) {
          atomic_max_f32(&out0[a], y);
        } else {
          float em = exps_max[a];
          atomic_add_f32(&out0[a], __expf(SMOOTH_F * (y - em)));
        }
      }
    }
  }
}

// ---------------- readout ----------------

__global__ void k_readout_sum(const float* __restrict__ h, const int* __restrict__ tok,
                              float* __restrict__ agg) {
  int b = blockIdx.x;
  long start = 0;
  for (int i = 0; i < b; ++i) start += tok[i];
  long end = start + tok[b];
  int t = threadIdx.x;           // 256 threads
  int col = t & 63, grp = t >> 6;
  float s = 0.0f;
  for (long r = start + grp; r < end; r += 4) s += h[r * EMB + col];
  __shared__ float red[256];
  red[t] = s;
  __syncthreads();
  if (grp == 0)
    agg[b * EMB + col] = red[col] + red[64 + col] + red[128 + col] + red[192 + col];
}

__global__ void k_readout_mlp(const float* __restrict__ agg,
                              const float* __restrict__ Wi, const float* __restrict__ bi,
                              const float* __restrict__ Wo, const float* __restrict__ bo,
                              float* __restrict__ out, int B) {
  int b = blockIdx.x * blockDim.x + threadIdx.x;
  if (b >= B) return;
  const float* x = agg + b * EMB;
  float o = bo[0];
  for (int j = 0; j < EMB; ++j) {
    float t = bi[j];
    for (int k = 0; k < EMB; ++k) t += x[k] * Wi[k * EMB + j];
    o += mish_f(t) * Wo[j];
  }
  out[b] = o;
}

// ---------------- host launcher ----------------

static inline size_t smem_sz(int WIN, int WHID, int WOUT) {
  return (size_t)(64 * WIN * 4 + 64 * WHID * 2 + WHID * WIN * 2 + WOUT * WHID * 2);
}

extern "C" void kernel_launch(void* const* d_in, const int* in_sizes, int n_in,
                              void* d_out, int out_size, void* d_ws, size_t ws_size,
                              hipStream_t stream) {
  const float* h0   = (const float*)d_in[0];
  const int* atoms1 = (const int*)d_in[1];
  const int* atoms2 = (const int*)d_in[2];
  const int* atoms3 = (const int*)d_in[3];
  const int* tok    = (const int*)d_in[4];
  const float* Wi1 = (const float*)d_in[5];  const float* bi1 = (const float*)d_in[6];
  const float* Wo1 = (const float*)d_in[7];  const float* bo1 = (const float*)d_in[8];
  const float* Wi2 = (const float*)d_in[9];  const float* bi2 = (const float*)d_in[10];
  const float* Wo2 = (const float*)d_in[11]; const float* bo2 = (const float*)d_in[12];
  const float* Wi3 = (const float*)d_in[13]; const float* bi3 = (const float*)d_in[14];
  const float* Wo3 = (const float*)d_in[15]; const float* bo3 = (const float*)d_in[16];
  const float* uWi = (const float*)d_in[17]; const float* ubi = (const float*)d_in[18];
  const float* uWo = (const float*)d_in[19]; const float* ubo = (const float*)d_in[20];
  const float* vWi = (const float*)d_in[21]; const float* vbi = (const float*)d_in[22];
  const float* vWo = (const float*)d_in[23]; const float* vbo = (const float*)d_in[24];
  const float* dWi = (const float*)d_in[25]; const float* dbi = (const float*)d_in[26];
  const float* dWo = (const float*)d_in[27]; const float* dbo = (const float*)d_in[28];

  const int N  = in_sizes[0] / EMB;
  const int m1 = in_sizes[1];
  const int m2 = in_sizes[2] / 2;
  const int m3 = in_sizes[3] / 3;
  const int B  = in_sizes[4];
  const size_t nf = (size_t)N * EMB;

  char* ws = (char*)d_ws;
  size_t off = 0;
  auto alloc = [&](size_t bytes) -> char* {
    char* p = ws + off;
    off += (bytes + 255) & ~(size_t)255;
    return p;
  };
  float* seg = (float*)alloc(nf * 4);
  float* sum = (float*)alloc(nf * 4);
  float* msg = (float*)alloc(nf * 4);
  float* h1  = (float*)alloc(nf * 4);
  float* h2  = (float*)alloc(nf * 4);
  float* agg = (float*)alloc((size_t)B * EMB * 4);
  unsigned short* wt_i1 = (unsigned short*)alloc(64 * 64 * 2);
  unsigned short* wt_o1 = (unsigned short*)alloc(64 * 64 * 2);
  unsigned short* wt_i2 = (unsigned short*)alloc(128 * 128 * 2);
  unsigned short* wt_o2 = (unsigned short*)alloc(128 * 128 * 2);
  unsigned short* wt_i3 = (unsigned short*)alloc(192 * 192 * 2);
  unsigned short* wt_o3 = (unsigned short*)alloc(192 * 192 * 2);
  unsigned short* wt_ui = (unsigned short*)alloc(128 * 128 * 2);
  unsigned short* wt_uo = (unsigned short*)alloc(128 * 64 * 2);
  (void)ws_size; (void)n_in; (void)out_size;

  auto conv = [&](const float* W, unsigned short* Wt, int K, int Nn) {
    int tot = K * Nn;
    k_convert_transpose<<<(tot + 255) / 256, 256, 0, stream>>>(W, Wt, K, Nn);
  };
  conv(Wi1, wt_i1, 64, 64);    conv(Wo1, wt_o1, 64, 64);
  conv(Wi2, wt_i2, 128, 128);  conv(Wo2, wt_o2, 128, 128);
  conv(Wi3, wt_i3, 192, 192);  conv(Wo3, wt_o3, 192, 192);
  conv(uWi, wt_ui, 128, 128);  conv(uWo, wt_uo, 128, 64);

  const int tot = (int)nf;
  const int g1 = (m1 + 63) / 64, g2 = (m2 + 63) / 64, g3 = (m3 + 63) / 64;
  const int gu = (N + 63) / 64;

  const float* hin = h0;
  float* houts[2] = {h1, h2};
  for (int layer = 0; layer < 2; ++layer) {
    k_init_agg<<<(tot + 255) / 256, 256, 0, stream>>>(seg, sum, tot);

    // pass A: smooth-max running max
    fused_gemm<64, 64, 64, 0, 1><<<g1, 128, smem_sz(64, 64, 64), stream>>>(
        hin, nullptr, atoms1, m1, wt_i1, bi1, wt_o1, bo1, seg, nullptr);
    fused_gemm<128, 128, 128, 0, 2><<<g2, 128, smem_sz(128, 128, 128), stream>>>(
        hin, nullptr, atoms2, m2, wt_i2, bi2, wt_o2, bo2, seg, nullptr);
    fused_gemm<192, 192, 192, 0, 3><<<g3, 128, smem_sz(192, 192, 192), stream>>>(
        hin, nullptr, atoms3, m3, wt_i3, bi3, wt_o3, bo3, seg, nullptr);

    k_fix_max<<<(tot + 255) / 256, 256, 0, stream>>>(seg, tot);

    // pass B: recompute messages, accumulate exp-sums
    fused_gemm<64, 64, 64, 1, 1><<<g1, 128, smem_sz(64, 64, 64), stream>>>(
        hin, nullptr, atoms1, m1, wt_i1, bi1, wt_o1, bo1, sum, seg);
    fused_gemm<128, 128, 128, 1, 2><<<g2, 128, smem_sz(128, 128, 128), stream>>>(
        hin, nullptr, atoms2, m2, wt_i2, bi2, wt_o2, bo2, sum, seg);
    fused_gemm<192, 192, 192, 1, 3><<<g3, 128, smem_sz(192, 192, 192), stream>>>(
        hin, nullptr, atoms3, m3, wt_i3, bi3, wt_o3, bo3, sum, seg);

    k_final_msg<<<(tot + 255) / 256, 256, 0, stream>>>(msg, sum, seg, tot);

    // update MLP
    fused_gemm<128, 128, 64, 2, 1><<<gu, 128, smem_sz(128, 128, 64), stream>>>(
        hin, msg, nullptr, N, wt_ui, ubi, wt_uo, ubo, houts[layer], nullptr);
    hin = houts[layer];
  }

  k_readout_sum<<<B, 256, 0, stream>>>(hin, tok, agg);
  k_readout_mlp<<<1, 64, 0, stream>>>(agg, vWi, vbi, vWo, vbo, (float*)d_out, B);
  k_readout_mlp<<<1, 64, 0, stream>>>(agg, dWi, dbi, dWo, dbo, (float*)d_out + B, B);
}
